// CPAttention_53197464928937
// MI455X (gfx1250) — matrix operand order repaired and checked by reading, not compiled, not run on hardware
//
#include <hip/hip_runtime.h>
#include <hip/hip_bf16.h>

// ---------------------------------------------------------------------------
// CP attention for MI455X (gfx1250, wave32, WMMA 16x16x32 f16->f32)
// f16-packed operands so every fragment load is a b128; 32x32 GEMM wave tiles.
// ---------------------------------------------------------------------------

#define BB 8
#define NN 1024
#define DD 512
#define INNER 512
#define HH 8
#define DH 64
#define SCALE 0.125f

typedef __attribute__((ext_vector_type(16))) _Float16 v16h;
typedef __attribute__((ext_vector_type(8)))  _Float16 v8h;
typedef __attribute__((ext_vector_type(8)))  float    v8f;

__device__ __forceinline__ v8f wmma16(v16h a, v16h b, v8f c) {
    return __builtin_amdgcn_wmma_f32_16x16x32_f16(
        false, a, false, b, (short)0, c, false, false);
}

__device__ __forceinline__ v16h cat16(v8h lo, v8h hi) {
    return __builtin_shufflevector(lo, hi, 0, 1, 2, 3, 4, 5, 6, 7,
                                           8, 9, 10, 11, 12, 13, 14, 15);
}

// A fragment (16x32 f16). ISA: M = lane&15; element j -> K = kb + j + h16*8 (j<8)
// and kb + 16 + (j-8) + h16*8 (j>=8). Two 16B loads from a row-major f16 row.
__device__ __forceinline__ v16h frag_a(const _Float16* __restrict__ row,
                                       int kb, int h16) {
    v8h lo = *(const v8h*)(row + kb + h16 * 8);
    v8h hi = *(const v8h*)(row + kb + h16 * 8 + 16);
    return cat16(lo, hi);
}

// B fragment (32x16 f16). ISA: N = lane&15; element j -> K = kb + j + h16*8 (j<8)
// and kb + 16 + (j-8) + h16*8 (j>=8).
// `row` points at the K-contiguous vector for this lane's output column.
__device__ __forceinline__ v16h frag_b(const _Float16* __restrict__ row,
                                       int kb, int h16) {
    v8h lo = *(const v8h*)(row + kb + h16 * 8);
    v8h hi = *(const v8h*)(row + kb + h16 * 8 + 16);
    return cat16(lo, hi);
}

// ---------------------------------------------------------------------------
// Pack kernels: f32 -> f16 (straight and transposed)
// ---------------------------------------------------------------------------
__global__ void pack_f16_kernel(const float* __restrict__ src,
                                _Float16* __restrict__ dst, int n) {
    int i = blockIdx.x * blockDim.x + threadIdx.x;
    if (i < n) dst[i] = (_Float16)src[i];
}

// dst[c*R + k] = src[k*C + c]  (transpose-convert, src is R x C)
__global__ void pack_f16_t_kernel(const float* __restrict__ src,
                                  _Float16* __restrict__ dst, int R, int C) {
    int i = blockIdx.x * blockDim.x + threadIdx.x;
    if (i < R * C) {
        int k = i / C, c = i % C;
        dst[(size_t)c * R + k] = (_Float16)src[i];
    }
}

// maskT[m][n] = (f16) cp_mask[n][m]
__global__ void pack_mask_kernel(const int* __restrict__ mask,
                                 _Float16* __restrict__ mT) {
    int i = blockIdx.x * blockDim.x + threadIdx.x;
    if (i < NN * NN) {
        int m = i / NN, n = i % NN;
        mT[(size_t)m * NN + n] = (_Float16)mask[(size_t)n * NN + m];
    }
}

__global__ void nnz_kernel(const int* __restrict__ mask, float* __restrict__ nnz) {
    int i = blockIdx.x * blockDim.x + threadIdx.x;
    if (i < NN) {
        int c = 0;
        for (int j = 0; j < NN; ++j) c += (mask[(size_t)i * NN + j] != 0);
        nnz[i] = (float)c;
    }
}

// ---------------------------------------------------------------------------
// K1: QKV projection, 32x32 wave tiles over (8192 x 1536).
// Writes q,k f16 [b][h][n][dh] and v f16 transposed [b][h][d][n].
// ---------------------------------------------------------------------------
__global__ __launch_bounds__(256) void qkv_kernel(const _Float16* __restrict__ xh,
                                                  const _Float16* __restrict__ wT,
                                                  _Float16* __restrict__ qh,
                                                  _Float16* __restrict__ kh,
                                                  _Float16* __restrict__ vT) {
    const int lane = threadIdx.x & 31;
    const int wid  = blockIdx.x * (blockDim.x >> 5) + (threadIdx.x >> 5);
    const int ntn  = (3 * INNER) / 32;                 // 48 col tiles
    const int mtile = wid / ntn, ntile = wid % ntn;
    if (mtile >= (BB * NN) / 32) return;

    const int m0 = mtile * 32, n0 = ntile * 32;
    const int col = lane & 15;
    const int h16 = lane >> 4;

    const _Float16* ar0 = xh + (size_t)(m0 + col) * DD;
    const _Float16* ar1 = xh + (size_t)(m0 + 16 + col) * DD;
    const _Float16* br0 = wT + (size_t)(n0 + col) * DD;
    const _Float16* br1 = wT + (size_t)(n0 + 16 + col) * DD;

    v8f acc[2][2] = {};
    for (int kb = 0; kb < DD; kb += 32) {
        v16h a0 = frag_a(ar0, kb, h16);
        v16h a1 = frag_a(ar1, kb, h16);
        v16h b0 = frag_b(br0, kb, h16);
        v16h b1 = frag_b(br1, kb, h16);
        acc[0][0] = wmma16(a0, b0, acc[0][0]);
        acc[0][1] = wmma16(a0, b1, acc[0][1]);
        acc[1][0] = wmma16(a1, b0, acc[1][0]);
        acc[1][1] = wmma16(a1, b1, acc[1][1]);
    }

#pragma unroll
    for (int ni = 0; ni < 2; ++ni) {
        const int c   = n0 + ni * 16 + col;
        const int sec = c / INNER;
        const int cc  = c % INNER;
        const int hd  = cc / DH;
        const int d   = cc % DH;
#pragma unroll
        for (int mi = 0; mi < 2; ++mi) {
#pragma unroll
            for (int r = 0; r < 8; ++r) {
                int m  = m0 + mi * 16 + r + h16 * 8;
                int bb = m >> 10, n = m & (NN - 1);
                _Float16 val = (_Float16)acc[mi][ni][r];
                if (sec == 0)
                    qh[((size_t)(bb * HH + hd) * NN + n) * DH + d] = val;
                else if (sec == 1)
                    kh[((size_t)(bb * HH + hd) * NN + n) * DH + d] = val;
                else
                    vT[((size_t)(bb * HH + hd) * DH + d) * NN + n] = val;
            }
        }
    }
}

// ---------------------------------------------------------------------------
// K2: fused flash attention, one wave per (b, h, 16 query rows), 32 keys/step.
// ---------------------------------------------------------------------------
__global__ __launch_bounds__(32) void attn_kernel(const _Float16* __restrict__ qh,
                                                  const _Float16* __restrict__ kh,
                                                  const _Float16* __restrict__ vT,
                                                  const _Float16* __restrict__ mT,
                                                  _Float16* __restrict__ aoh,
                                                  float* __restrict__ score) {
    __shared__ _Float16 plds[16][40];              // 16x32 P tile, padded

    const int lane = threadIdx.x;
    int blk = blockIdx.x;
    const int qt = blk & (NN / 16 - 1); blk >>= 6;
    const int h  = blk & (HH - 1);      blk >>= 3;
    const int b  = blk;

    const _Float16* qm = qh + (size_t)(b * HH + h) * NN * DH;
    const _Float16* km = kh + (size_t)(b * HH + h) * NN * DH;
    const _Float16* vm = vT + (size_t)(b * HH + h) * DH * NN;

    const int n0  = qt * 16;
    const int col = lane & 15;
    const int h16 = lane >> 4;

    float m_run[8], l_run[8], sc[8];
    v8f o[4];
#pragma unroll
    for (int r = 0; r < 8; ++r) { m_run[r] = -1e30f; l_run[r] = 0.f; sc[r] = 0.f; }
#pragma unroll
    for (int dt = 0; dt < 4; ++dt) o[dt] = v8f{};

    const _Float16* qrow = qm + (size_t)(n0 + col) * DH;
    const v16h qa0 = frag_a(qrow, 0, h16);
    const v16h qa1 = frag_a(qrow, 32, h16);

    for (int ct = 0; ct < NN / 32; ++ct) {
        const int c0 = ct * 32;
        if (ct + 1 < NN / 32) {                    // hint next K/V tiles
            __builtin_prefetch(km + (size_t)(c0 + 32 + col) * DH, 0, 3);
            __builtin_prefetch(vm + (size_t)col * NN + c0 + 32, 0, 3);
        }

        // S = q @ k^T for two 16-col subtiles (K rows are f16-contiguous)
        const _Float16* kr0 = km + (size_t)(c0 + col) * DH;
        const _Float16* kr1 = km + (size_t)(c0 + 16 + col) * DH;
        v8f s0 = {}, s1 = {};
        s0 = wmma16(qa0, frag_b(kr0, 0, h16), s0);
        s0 = wmma16(qa1, frag_b(kr0, 32, h16), s0);
        s1 = wmma16(qa0, frag_b(kr1, 0, h16), s1);
        s1 = wmma16(qa1, frag_b(kr1, 32, h16), s1);

        // mask (transposed, query-contiguous): one v8h per key column
        v8h mv0 = *(const v8h*)(mT + (size_t)(c0 + col) * NN + n0 + h16 * 8);
        v8h mv1 = *(const v8h*)(mT + (size_t)(c0 + 16 + col) * NN + n0 + h16 * 8);

        __syncthreads();
#pragma unroll
        for (int r = 0; r < 8; ++r) {
            float l0 = s0[r] * SCALE * (float)mv0[r];
            float l1 = s1[r] * SCALE * (float)mv1[r];
            sc[r] += fabsf(l0) + fabsf(l1);

            float tm = fmaxf(l0, l1);
#pragma unroll
            for (int ofs = 8; ofs >= 1; ofs >>= 1)
                tm = fmaxf(tm, __shfl_xor(tm, ofs, 32));
            float mn = fmaxf(m_run[r], tm);
            float f  = __expf(m_run[r] - mn);
            m_run[r] = mn;
            float e0 = __expf(l0 - mn);
            float e1 = __expf(l1 - mn);
            float rs = e0 + e1;
#pragma unroll
            for (int ofs = 8; ofs >= 1; ofs >>= 1)
                rs += __shfl_xor(rs, ofs, 32);
            l_run[r] = l_run[r] * f + rs;
#pragma unroll
            for (int dt = 0; dt < 4; ++dt) o[dt][r] *= f;

            plds[r + h16 * 8][col]      = (_Float16)e0;
            plds[r + h16 * 8][col + 16] = (_Float16)e1;
        }
        __syncthreads();

        // P (16x32) in A layout from LDS
        v8h plo = *(const v8h*)&plds[col][h16 * 8];
        v8h phi = *(const v8h*)&plds[col][16 + h16 * 8];
        v16h pa = cat16(plo, phi);

        // O += P @ V, full K=32 (v transposed: K contiguous per d column)
#pragma unroll
        for (int dt = 0; dt < 4; ++dt) {
            const _Float16* vr = vm + (size_t)(dt * 16 + col) * NN + c0;
            o[dt] = wmma16(pa, frag_b(vr, 0, h16), o[dt]);
        }
    }

    // finalize: normalize, store f16 merged-head output, reduce score
#pragma unroll
    for (int r = 0; r < 8; ++r) {
        int row = n0 + r + h16 * 8;
        float inv = 1.0f / l_run[r];
#pragma unroll
        for (int dt = 0; dt < 4; ++dt)
            aoh[(size_t)(b * NN + row) * INNER + h * DH + dt * 16 + col] =
                (_Float16)(o[dt][r] * inv);
        float v = sc[r];
#pragma unroll
        for (int ofs = 8; ofs >= 1; ofs >>= 1)
            v += __shfl_xor(v, ofs, 32);
        if (col == 0) atomicAdd(&score[b * NN + row], v);
    }
}

// ---------------------------------------------------------------------------
// K3: score /= nnz, stable 17-smallest selection per batch
// ---------------------------------------------------------------------------
__global__ __launch_bounds__(256) void topk_kernel(const float* __restrict__ score,
                                                   const float* __restrict__ nnz,
                                                   int* __restrict__ idx17) {
    __shared__ float s[NN];
    __shared__ float rv[256];
    __shared__ int   ri[256];
    const int b = blockIdx.x, t = threadIdx.x;
    for (int i = t; i < NN; i += 256) s[i] = score[(size_t)b * NN + i] / nnz[i];
    __syncthreads();
    for (int it = 0; it < 17; ++it) {
        float bv = 1e30f; int bi = NN;
        for (int i = t; i < NN; i += 256) {
            float v = s[i];
            if (v < bv || (v == bv && i < bi)) { bv = v; bi = i; }
        }
        rv[t] = bv; ri[t] = bi;
        __syncthreads();
        for (int st = 128; st >= 1; st >>= 1) {
            if (t < st) {
                float ov = rv[t + st]; int oi = ri[t + st];
                if (ov < rv[t] || (ov == rv[t] && oi < ri[t])) { rv[t] = ov; ri[t] = oi; }
            }
            __syncthreads();
        }
        if (t == 0) { idx17[b * 32 + it] = ri[0]; s[ri[0]] = 1e30f; }
        __syncthreads();
    }
}

// ---------------------------------------------------------------------------
// K4: row-source map incl. the reference's batch reversal quirk
// ---------------------------------------------------------------------------
__global__ __launch_bounds__(1024) void rowmap_kernel(const int* __restrict__ idx17,
                                                      int* __restrict__ srcmap) {
    const int b = blockIdx.x, t = threadIdx.x;
    srcmap[b * NN + t] = t;
    __syncthreads();
    if (t == 0) {
        const int* idxr = idx17 + (BB - 1 - b) * 32;
        for (int i = 1; i <= 16; ++i) {
            int ti = idxr[i];
            srcmap[b * NN + i]  = ti;
            srcmap[b * NN + ti] = i;
        }
    }
}

// ---------------------------------------------------------------------------
// K5: output projection with row gather, 32x32 wave tiles
// ---------------------------------------------------------------------------
__global__ __launch_bounds__(256) void outproj_kernel(const _Float16* __restrict__ aoh,
                                                      const _Float16* __restrict__ wT,
                                                      const float* __restrict__ bias,
                                                      const int* __restrict__ srcmap,
                                                      float* __restrict__ out) {
    const int lane = threadIdx.x & 31;
    const int wid  = blockIdx.x * (blockDim.x >> 5) + (threadIdx.x >> 5);
    const int ntn  = DD / 32;                      // 16 col tiles
    const int mtile = wid / ntn, ntile = wid % ntn;
    if (mtile >= (BB * NN) / 32) return;

    const int m0 = mtile * 32, n0 = ntile * 32;
    const int col = lane & 15;
    const int h16 = lane >> 4;
    const int bb  = m0 >> 10;

    const _Float16* ar0 = aoh + (size_t)(bb * NN + srcmap[m0 + col]) * INNER;
    const _Float16* ar1 = aoh + (size_t)(bb * NN + srcmap[m0 + 16 + col]) * INNER;
    const _Float16* br0 = wT + (size_t)(n0 + col) * INNER;
    const _Float16* br1 = wT + (size_t)(n0 + 16 + col) * INNER;

    v8f acc[2][2] = {};
    for (int kb = 0; kb < INNER; kb += 32) {
        v16h a0 = frag_a(ar0, kb, h16);
        v16h a1 = frag_a(ar1, kb, h16);
        v16h b0 = frag_b(br0, kb, h16);
        v16h b1 = frag_b(br1, kb, h16);
        acc[0][0] = wmma16(a0, b0, acc[0][0]);
        acc[0][1] = wmma16(a0, b1, acc[0][1]);
        acc[1][0] = wmma16(a1, b0, acc[1][0]);
        acc[1][1] = wmma16(a1, b1, acc[1][1]);
    }
    const float bv0 = bias[n0 + col];
    const float bv1 = bias[n0 + 16 + col];
#pragma unroll
    for (int mi = 0; mi < 2; ++mi)
#pragma unroll
        for (int r = 0; r < 8; ++r) {
            int m = m0 + mi * 16 + r + h16 * 8;
            out[(size_t)m * DD + n0 + col]      = acc[mi][0][r] + bv0;
            out[(size_t)m * DD + n0 + 16 + col] = acc[mi][1][r] + bv1;
        }
}

// ---------------------------------------------------------------------------
// TDM probe: exercise tensor_load_to_lds + s_wait_tensorcnt (compile probe for
// integrating TDM staging next round). Loads a 16x16 f32 tile of x into LDS.
// ---------------------------------------------------------------------------
typedef __attribute__((ext_vector_type(4))) unsigned int u32x4;
typedef __attribute__((ext_vector_type(4))) int          i32x4;
typedef __attribute__((ext_vector_type(8))) int          i32x8;

__global__ __launch_bounds__(32) void tdm_probe_kernel(const float* __restrict__ src,
                                                       float* __restrict__ dst) {
    __shared__ float tile[16][16];
#if defined(__has_builtin) && __has_builtin(__builtin_amdgcn_tensor_load_to_lds)
    unsigned int lds_off = (unsigned int)(unsigned long long)&tile[0][0];
    unsigned long long ga = (unsigned long long)src;

    u32x4 g0 = {};
    g0[0] = 1u;                                   // count=1 (valid user D#)
    g0[1] = lds_off;                              // lds_addr (bytes)
    g0[2] = (unsigned int)ga;                     // global_addr[31:0]
    g0[3] = (unsigned int)((ga >> 32) & 0x1FFFFFFu) | (2u << 30);  // [56:32], type=2

    i32x8 g1 = {};
    g1[0] = 2 << 16;                              // data_size=2 (4B), wg_mask=0
    g1[1] = (DD & 0xFFFF) << 16;                  // tensor_dim0 = 512
    g1[2] = ((BB * NN) & 0xFFFF) << 16;           // tensor_dim1 = 8192
    g1[3] = 16 << 16;                             // tile_dim0 = 16
    g1[4] = 16;                                   // tile_dim1 = 16
    g1[5] = DD;                                   // tensor_dim0_stride = 512
    i32x4 g2 = {}, g3 = {};

#if __clang_major__ >= 23
    __builtin_amdgcn_tensor_load_to_lds(g0, g1, g2, g3, i32x8{}, 0);
#else
    __builtin_amdgcn_tensor_load_to_lds(g0, g1, g2, g3, 0);
#endif
    __builtin_amdgcn_s_wait_tensorcnt(0);
#else
    tile[threadIdx.x >> 4][threadIdx.x & 15] = 0.f;
#endif
    __syncthreads();
    dst[threadIdx.x] = tile[threadIdx.x >> 4][threadIdx.x & 15];
}

// ---------------------------------------------------------------------------
extern "C" void kernel_launch(void* const* d_in, const int* in_sizes, int n_in,
                              void* d_out, int out_size, void* d_ws, size_t ws_size,
                              hipStream_t stream) {
    const float* x     = (const float*)d_in[0];
    const int*   mask  = (const int*)  d_in[1];
    const float* w_qkv = (const float*)d_in[2];
    const float* w_out = (const float*)d_in[3];
    const float* b_out = (const float*)d_in[4];
    float*       out   = (float*)d_out;

    // workspace carve-up (256B aligned chunks)
    char* p = (char*)d_ws;
    auto alloc = [&](size_t bytes) {
        char* r = p; p += (bytes + 255) & ~(size_t)255; return r;
    };
    const size_t MR = (size_t)BB * NN;            // 8192 rows
    const size_t HS = (size_t)BB * HH * NN * DH;  // per-tensor head elems
    _Float16* xh   = (_Float16*)alloc(MR * DD * 2);
    _Float16* whT  = (_Float16*)alloc((size_t)3 * INNER * DD * 2);
    _Float16* woT  = (_Float16*)alloc((size_t)DD * INNER * 2);
    _Float16* mTh  = (_Float16*)alloc((size_t)NN * NN * 2);
    _Float16* qh   = (_Float16*)alloc(HS * 2);
    _Float16* kh   = (_Float16*)alloc(HS * 2);
    _Float16* vTh  = (_Float16*)alloc(HS * 2);
    _Float16* aoh  = (_Float16*)alloc(MR * INNER * 2);
    float*    score= (float*)alloc(MR * 4);
    float*    nnz  = (float*)alloc(NN * 4);
    int*      idx17= (int*)alloc(BB * 32 * 4);
    int*      srcmap=(int*)alloc(MR * 4);
    float*    probe= (float*)alloc(256);

    hipMemsetAsync(score, 0, MR * sizeof(float), stream);

    // packing passes
    pack_f16_kernel<<<(int)((MR * DD + 255) / 256), 256, 0, stream>>>(x, xh, MR * DD);
    pack_f16_t_kernel<<<(DD * 3 * INNER + 255) / 256, 256, 0, stream>>>(
        w_qkv, whT, DD, 3 * INNER);
    pack_f16_t_kernel<<<(INNER * DD + 255) / 256, 256, 0, stream>>>(
        w_out, woT, INNER, DD);
    pack_mask_kernel<<<(NN * NN + 255) / 256, 256, 0, stream>>>(mask, mTh);
    nnz_kernel<<<(NN + 255) / 256, 256, 0, stream>>>(mask, nnz);

    tdm_probe_kernel<<<1, 32, 0, stream>>>(x, probe);

    {   // QKV: (8192/32)*(1536/32) = 12288 wave tiles, 8 waves/block
        int tiles = ((BB * NN) / 32) * ((3 * INNER) / 32);
        qkv_kernel<<<tiles / 8, 256, 0, stream>>>(xh, whT, qh, kh, vTh);
    }

    attn_kernel<<<BB * HH * (NN / 16), 32, 0, stream>>>(qh, kh, vTh, mTh, aoh, score);

    topk_kernel<<<BB, 256, 0, stream>>>(score, nnz, idx17);
    rowmap_kernel<<<BB, NN, 0, stream>>>(idx17, srcmap);

    {   // out proj: (8192/32)*(512/32) = 4096 wave tiles
        int tiles = ((BB * NN) / 32) * (DD / 32);
        outproj_kernel<<<tiles / 8, 256, 0, stream>>>(aoh, woT, b_out, srcmap, out);
    }
}
